// encoder_51651276702218
// MI455X (gfx1250) — matrix operand order repaired and checked by reading, not compiled, not run on hardware
//
#include <hip/hip_runtime.h>
#include <hip/hip_bf16.h>

typedef _Float16 half_t;
typedef __attribute__((ext_vector_type(16))) _Float16 v16h;
typedef __attribute__((ext_vector_type(8)))  float    v8f;

#define SEQ_N 2048
#define DIM   512
#define HEADS 8
#define DEPTH 64
#define DFF   2048
#define BATCH 4

__device__ __forceinline__ v8f wmma_f16(v16h a, v16h b, v8f c) {
  return __builtin_amdgcn_wmma_f32_16x16x32_f16(false, a, false, b, (short)0, c, false, false);
}

// ---------------------------------------------------------------------------
// CDNA5 async global->LDS copy (ASYNCcnt-tracked, ISA §15.18.3 op 98).
// LDS address = low 32 bits of the flat pointer (ISA §10.2: LDS aperture
// addresses carry the wave-relative LDS byte address in addr[31:0]).
// Requires 16B-aligned LDS destination.
// ---------------------------------------------------------------------------
__device__ __forceinline__ void async_ld_b128(void* lds_dst, const void* gsrc) {
  unsigned loff = (unsigned)(unsigned long long)lds_dst;
  asm volatile("global_load_async_to_lds_b128 %0, %1, off"
               :: "v"(loff), "v"(gsrc)
               : "memory");
}
__device__ __forceinline__ void wait_async0() {
  asm volatile("s_wait_asynccnt 0x0" ::: "memory");
}

// ---------------------------------------------------------------------------
// fp32 -> f16 conversion
// ---------------------------------------------------------------------------
__global__ void cvt_f16_kernel(const float* __restrict__ in, half_t* __restrict__ out, int n) {
  int i = blockIdx.x * blockDim.x + threadIdx.x;
  if (i < n) out[i] = (half_t)in[i];
}

// ---------------------------------------------------------------------------
// WMMA GEMM: out = epilogue(A[MxK]_f16 @ B[KxN]_f16 + bias), fp32 accum.
// Block tile 128x64, K-step 64, double-buffered software pipeline:
//   iter i: issue async A / transpose B for tile i+1 -> nxt buffer,
//           run 8 WMMAs on cur buffer, then s_wait_asynccnt + barrier.
// 8 waves of 32 (wave32), each wave computes 32x32.
// Epilogue: +bias[n]; optional ReLU; *scale; optional +res; store f32/f16.
// M%128==0, N%64==0, K%64==0 (true for all uses here).
// ---------------------------------------------------------------------------
#define ASTR 80   // async-staged tile row stride (160B, 16B-aligned rows)
#define BSTR 72

__global__ __launch_bounds__(256)
void gemm_f16_kernel(const half_t* __restrict__ A, const half_t* __restrict__ B,
                     const float* __restrict__ bias, const float* __restrict__ res,
                     float* __restrict__ out32, half_t* __restrict__ out16,
                     int M, int N, int K, float scale, int relu)
{
  __shared__ half_t a_lds[2][128 * ASTR];   // [m][k]
  __shared__ half_t b_lds[2][64 * BSTR];    // transposed: [n][k]

  const int t = threadIdx.x;
  const int wave = t >> 5, lane = t & 31, lo = lane & 15, hi = lane >> 4;
  const int wm = (wave & 3) * 32, wn = (wave >> 2) * 32;
  const int m0 = blockIdx.y * 128, n0 = blockIdx.x * 64;

  auto stage_a = [&](int buf, int k0) {   // 128x64 halves via async copy
    #pragma unroll
    for (int i = 0; i < 4; ++i) {
      int c = t + 256 * i;
      int r = c >> 3, kk = (c & 7) * 8;
      async_ld_b128(&a_lds[buf][r * ASTR + kk], &A[(size_t)(m0 + r) * K + k0 + kk]);
    }
  };
  auto stage_b = [&](int buf, int k0) {   // 64x64, transposed into [n][k]
    #pragma unroll
    for (int i = 0; i < 2; ++i) {
      int c = t + 256 * i;
      int kk = c >> 3, nb = (c & 7) * 8;
      uint4 d = *(const uint4*)&B[(size_t)(k0 + kk) * N + n0 + nb];
      const half_t* hh = (const half_t*)&d;
      #pragma unroll
      for (int j = 0; j < 8; ++j) b_lds[buf][(nb + j) * BSTR + kk] = hh[j];
    }
  };

  v8f acc[2][2] = {};

  // prologue: stage tile 0
  stage_a(0, 0);
  stage_b(0, 0);
  wait_async0();
  __syncthreads();

  int cur = 0;
  for (int k0 = 0; k0 < K; k0 += 64) {
    const int nxt = cur ^ 1;
    const bool more = (k0 + 64 < K);
    if (more) {                       // stage tile i+1 while computing tile i
      stage_a(nxt, k0 + 64);
      stage_b(nxt, k0 + 64);
    }
    if (k0 + 128 < K) {               // warm GL2 two tiles ahead
      int kk = t >> 3, nb = (t & 7) * 8;
      __builtin_prefetch(&B[(size_t)(k0 + 128 + kk) * N + n0 + nb], 0, 0);
    }

    #pragma unroll
    for (int ks = 0; ks < 2; ++ks) {
      v16h af[2], bf[2];
      #pragma unroll
      for (int i = 0; i < 2; ++i) {
        union { v16h h; unsigned u[8]; } f;
        #pragma unroll
        for (int v = 0; v < 8; ++v) {
          int k = ks * 32 + 2 * v + ((v & 4) ? 8 : 0) + hi * 8;      // ISA A-frag layout
          f.u[v] = *(const unsigned*)&a_lds[cur][(wm + i * 16 + lo) * ASTR + k];
        }
        af[i] = f.h;
      }
      #pragma unroll
      for (int j = 0; j < 2; ++j) {
        union { v16h h; unsigned u[8]; } f;
        #pragma unroll
        for (int v = 0; v < 8; ++v) {
          int k = ks * 32 + 2 * v + ((v & 4) ? 8 : 0) + hi * 8;      // ISA B-frag layout
          f.u[v] = *(const unsigned*)&b_lds[cur][(wn + j * 16 + lo) * BSTR + k];
        }
        bf[j] = f.h;
      }
      #pragma unroll
      for (int i = 0; i < 2; ++i)
        #pragma unroll
        for (int j = 0; j < 2; ++j)
          acc[i][j] = wmma_f16(af[i], bf[j], acc[i][j]);
    }

    if (more) {
      wait_async0();                  // tile i+1's async copies (overlapped above)
      __syncthreads();
      cur = nxt;
    }
  }

  #pragma unroll
  for (int i = 0; i < 2; ++i)
    #pragma unroll
    for (int j = 0; j < 2; ++j)
      #pragma unroll
      for (int v = 0; v < 8; ++v) {
        int r = m0 + wm + i * 16 + v + hi * 8;   // ISA C-frag layout
        int c = n0 + wn + j * 16 + lo;
        float val = acc[i][j][v] + bias[c];
        if (relu) val = fmaxf(val, 0.0f);
        val *= scale;
        size_t idx = (size_t)r * N + c;
        if (res)   val += res[idx];
        if (out32) out32[idx] = val;
        if (out16) out16[idx] = (half_t)val;
      }
}

// ---------------------------------------------------------------------------
// Flash attention: Z[b,n,h,d] = softmax(Q K^T) V.  Q pre-scaled by 1/8.
// grid = (B*H, N/128); 8 waves, each wave owns 16 query rows, loops over
// 64-key blocks double-buffered in LDS (K via async copy, V transposed by
// VALU) with online softmax (fp32 stats in registers).
// ---------------------------------------------------------------------------
#define KSTR 80   // async-staged K tile (16B-aligned rows)
#define DSTR 72

__global__ __launch_bounds__(256)
void attn_kernel(const half_t* __restrict__ Q, const half_t* __restrict__ Km,
                 const half_t* __restrict__ V, half_t* __restrict__ Z)
{
  __shared__ half_t kt[2][64 * KSTR];     // [key][d]   (B-frag source for S)
  __shared__ half_t vt[2][64 * DSTR];     // [d][key]   (B-frag source for O)
  __shared__ half_t pscr[8][16 * DSTR];   // per-wave P re-layout scratch

  const int t = threadIdx.x;
  const int wave = t >> 5, lane = t & 31, lo = lane & 15, hi = lane >> 4;
  const int bh = blockIdx.x;
  const int b = bh >> 3, h = bh & 7;
  const int q0 = blockIdx.y * 128 + wave * 16;
  const size_t base = (size_t)b * SEQ_N;

  auto stage_kv = [&](int buf, int kb) {
    #pragma unroll
    for (int i = 0; i < 2; ++i) {     // K block: straight async copy
      int c = t + 256 * i;
      int key = c >> 3, dd = (c & 7) * 8;
      async_ld_b128(&kt[buf][key * KSTR + dd],
                    &Km[(base + kb + key) * DIM + h * DEPTH + dd]);
    }
    #pragma unroll
    for (int i = 0; i < 2; ++i) {     // V block: transpose into [d][key]
      int c = t + 256 * i;
      int key = c >> 3, db = (c & 7) * 8;
      uint4 d = *(const uint4*)&V[(base + kb + key) * DIM + h * DEPTH + db];
      const half_t* hh = (const half_t*)&d;
      #pragma unroll
      for (int j = 0; j < 8; ++j) vt[buf][(db + j) * DSTR + key] = hh[j];
    }
  };

  // Q A-fragments (held for the whole kernel)
  v16h aq[2];
  #pragma unroll
  for (int jj = 0; jj < 2; ++jj) {
    union { v16h hh; unsigned u[8]; } f;
    #pragma unroll
    for (int v = 0; v < 8; ++v) {
      int k = jj * 32 + 2 * v + ((v & 4) ? 8 : 0) + hi * 8;
      f.u[v] = *(const unsigned*)&Q[(base + q0 + lo) * DIM + h * DEPTH + k];
    }
    aq[jj] = f.hh;
  }

  float mrow[8], lrow[8];
  #pragma unroll
  for (int v = 0; v < 8; ++v) { mrow[v] = -1e30f; lrow[v] = 0.0f; }
  v8f o[4] = {};

  // prologue: stage key block 0
  stage_kv(0, 0);
  wait_async0();
  __syncthreads();

  int cur = 0;
  for (int kb = 0; kb < SEQ_N; kb += 64) {
    const int nxt = cur ^ 1;
    const bool more = (kb + 64 < SEQ_N);
    if (more) stage_kv(nxt, kb + 64);   // overlap DMA with this block's math

    // S = Q K^T  (16 x 64 per wave, four 16x16 fragments)
    v8f s[4];
    #pragma unroll
    for (int f = 0; f < 4; ++f) {
      v8f c = {};
      #pragma unroll
      for (int jj = 0; jj < 2; ++jj) {
        union { v16h hh; unsigned u[8]; } bf;
        #pragma unroll
        for (int v = 0; v < 8; ++v) {
          int k = jj * 32 + 2 * v + ((v & 4) ? 8 : 0) + hi * 8;
          bf.u[v] = *(const unsigned*)&kt[cur][(f * 16 + lo) * KSTR + k];
        }
        c = wmma_f16(aq[jj], bf.hh, c);
      }
      s[f] = c;
    }

    // online softmax (fp32)
    float mnew[8], alpha[8], rs[8];
    #pragma unroll
    for (int v = 0; v < 8; ++v) {
      float mx = fmaxf(fmaxf(s[0][v], s[1][v]), fmaxf(s[2][v], s[3][v]));
      #pragma unroll
      for (int off = 8; off > 0; off >>= 1) mx = fmaxf(mx, __shfl_xor(mx, off, 32));
      mnew[v]  = fmaxf(mrow[v], mx);
      alpha[v] = __expf(mrow[v] - mnew[v]);
      mrow[v]  = mnew[v];
      rs[v] = 0.0f;
    }
    #pragma unroll
    for (int f = 0; f < 4; ++f)
      #pragma unroll
      for (int v = 0; v < 8; ++v) {
        float p = __expf(s[f][v] - mnew[v]);
        s[f][v] = p;
        rs[v] += p;
      }
    #pragma unroll
    for (int v = 0; v < 8; ++v) {
      #pragma unroll
      for (int off = 8; off > 0; off >>= 1) rs[v] += __shfl_xor(rs[v], off, 32);
      lrow[v] = lrow[v] * alpha[v] + rs[v];
    }
    #pragma unroll
    for (int f = 0; f < 4; ++f)
      #pragma unroll
      for (int v = 0; v < 8; ++v) o[f][v] *= alpha[v];

    // P (C-layout) -> f16 A-layout via per-wave LDS scratch (same-wave, DS in-order)
    #pragma unroll
    for (int f = 0; f < 4; ++f)
      #pragma unroll
      for (int v = 0; v < 8; ++v)
        pscr[wave][(v + hi * 8) * DSTR + f * 16 + lo] = (half_t)s[f][v];

    v16h ap[2];
    #pragma unroll
    for (int jj = 0; jj < 2; ++jj) {
      union { v16h hh; unsigned u[8]; } f;
      #pragma unroll
      for (int v = 0; v < 8; ++v) {
        int k = jj * 32 + 2 * v + ((v & 4) ? 8 : 0) + hi * 8;
        f.u[v] = *(const unsigned*)&pscr[wave][lo * DSTR + k];
      }
      ap[jj] = f.hh;
    }

    // O += P V
    #pragma unroll
    for (int f = 0; f < 4; ++f)
      #pragma unroll
      for (int jj = 0; jj < 2; ++jj) {
        union { v16h hh; unsigned u[8]; } bf;
        #pragma unroll
        for (int v = 0; v < 8; ++v) {
          int k = jj * 32 + 2 * v + ((v & 4) ? 8 : 0) + hi * 8;
          bf.u[v] = *(const unsigned*)&vt[cur][(f * 16 + lo) * DSTR + k];
        }
        o[f] = wmma_f16(ap[jj], bf.hh, o[f]);
      }

    if (more) {
      wait_async0();                  // next block's async K copies
      __syncthreads();
      cur = nxt;
    }
  }

  // normalize and store Z in (b, n, h*64+d) layout, f16
  #pragma unroll
  for (int f = 0; f < 4; ++f)
    #pragma unroll
    for (int v = 0; v < 8; ++v) {
      float val = o[f][v] / lrow[v];
      Z[(base + q0 + v + hi * 8) * DIM + h * DEPTH + f * 16 + lo] = (half_t)val;
    }
}

// ---------------------------------------------------------------------------
// LayerNorm over (N, D) per batch sample: deterministic two-stage reduction.
// ---------------------------------------------------------------------------
#define LN_PARTS 64

__global__ __launch_bounds__(256)
void ln_partial_kernel(const float* __restrict__ src, float* __restrict__ part, int perBatch)
{
  const int b = blockIdx.y, p = blockIdx.x;
  const int chunk = perBatch / LN_PARTS;
  const float* s = src + (size_t)b * perBatch + (size_t)p * chunk;
  float sum = 0.0f, sq = 0.0f;
  for (int i = threadIdx.x; i < chunk; i += 256) {
    float v = s[i];
    sum += v; sq += v * v;
  }
  #pragma unroll
  for (int off = 16; off > 0; off >>= 1) {
    sum += __shfl_xor(sum, off, 32);
    sq  += __shfl_xor(sq, off, 32);
  }
  __shared__ float red[2][8];
  int wave = threadIdx.x >> 5;
  if ((threadIdx.x & 31) == 0) { red[0][wave] = sum; red[1][wave] = sq; }
  __syncthreads();
  if (threadIdx.x == 0) {
    float S = 0.0f, SQ = 0.0f;
    #pragma unroll
    for (int i = 0; i < 8; ++i) { S += red[0][i]; SQ += red[1][i]; }
    part[(b * LN_PARTS + p) * 2 + 0] = S;
    part[(b * LN_PARTS + p) * 2 + 1] = SQ;
  }
}

__global__ void ln_finalize_kernel(const float* __restrict__ part, float* __restrict__ stats,
                                   float invCount)
{
  const int b = blockIdx.x;
  float s = 0.0f, sq = 0.0f;
  for (int i = threadIdx.x; i < LN_PARTS; i += 32) {
    s  += part[(b * LN_PARTS + i) * 2 + 0];
    sq += part[(b * LN_PARTS + i) * 2 + 1];
  }
  #pragma unroll
  for (int off = 16; off > 0; off >>= 1) {
    s  += __shfl_xor(s, off, 32);
    sq += __shfl_xor(sq, off, 32);
  }
  if (threadIdx.x == 0) {
    float mu  = s * invCount;
    float var = sq * invCount - mu * mu;
    stats[b * 2 + 0] = mu;
    stats[b * 2 + 1] = rsqrtf(var + 1e-6f);
  }
}

__global__ void ln_apply_kernel(const float* __restrict__ src, const float* __restrict__ g,
                                const float* __restrict__ be, const float* __restrict__ stats,
                                float* __restrict__ o32, half_t* __restrict__ o16, int perBatch)
{
  size_t idx = (size_t)blockIdx.x * blockDim.x + threadIdx.x;
  int b = (int)(idx / perBatch);
  int e = (int)(idx % perBatch);
  float mu = stats[b * 2], rs = stats[b * 2 + 1];
  float val = (src[idx] - mu) * rs * g[e] + be[e];
  if (o32) o32[idx] = val;
  if (o16) o16[idx] = (half_t)val;
}

// ---------------------------------------------------------------------------
// Host side
// ---------------------------------------------------------------------------
static void launch_cvt(const float* in, half_t* out, int n, hipStream_t s) {
  cvt_f16_kernel<<<(n + 255) / 256, 256, 0, s>>>(in, out, n);
}

static void launch_gemm(const half_t* A, const half_t* B, const float* bias,
                        const float* res, float* o32, half_t* o16,
                        int M, int N, int K, float scale, int relu, hipStream_t s) {
  dim3 g(N / 64, M / 128);
  gemm_f16_kernel<<<g, 256, 0, s>>>(A, B, bias, res, o32, o16, M, N, K, scale, relu);
}

extern "C" void kernel_launch(void* const* d_in, const int* in_sizes, int n_in,
                              void* d_out, int out_size, void* d_ws, size_t ws_size,
                              hipStream_t stream) {
  const float* x  = (const float*)d_in[0];
  const float* Wq = (const float*)d_in[1];  const float* bq = (const float*)d_in[2];
  const float* Wk = (const float*)d_in[3];  const float* bk = (const float*)d_in[4];
  const float* Wv = (const float*)d_in[5];  const float* bv = (const float*)d_in[6];
  const float* Wo = (const float*)d_in[7];  const float* bo = (const float*)d_in[8];
  const float* W1 = (const float*)d_in[9];  const float* b1 = (const float*)d_in[10];
  const float* W2 = (const float*)d_in[11]; const float* b2 = (const float*)d_in[12];
  const float* g1 = (const float*)d_in[13]; const float* be1 = (const float*)d_in[14];
  const float* g2 = (const float*)d_in[15]; const float* be2 = (const float*)d_in[16];

  const int M  = BATCH * SEQ_N;        // 8192
  const int PB = SEQ_N * DIM;          // 1048576 elems per batch sample
  const int TOT = BATCH * PB;          // 4194304

  // workspace bump allocator (all sizes multiples of 256B)
  char* p = (char*)d_ws;
  auto take = [&](size_t bytes) { char* r = p; p += (bytes + 255) & ~(size_t)255; return r; };

  half_t* x16  = (half_t*)take((size_t)TOT * 2);
  half_t* wq16 = (half_t*)take((size_t)DIM * DIM * 2);
  half_t* wk16 = (half_t*)take((size_t)DIM * DIM * 2);
  half_t* wv16 = (half_t*)take((size_t)DIM * DIM * 2);
  half_t* wo16 = (half_t*)take((size_t)DIM * DIM * 2);
  half_t* w1_16 = (half_t*)take((size_t)DIM * DFF * 2);
  half_t* w2_16 = (half_t*)take((size_t)DFF * DIM * 2);
  half_t* q16  = (half_t*)take((size_t)TOT * 2);          // |
  half_t* k16  = (half_t*)take((size_t)TOT * 2);          // | contiguous: reused as
  half_t* v16  = (half_t*)take((size_t)TOT * 2);          // | ffn hidden (8192x2048 f16)
  half_t* z16  = (half_t*)take((size_t)TOT * 2);          // |
  float*  hbuf = (float*)take((size_t)TOT * 4);           // pre-LN sums (reused for LN2)
  float*  out1 = (float*)take((size_t)TOT * 4);
  half_t* out1_16 = (half_t*)take((size_t)TOT * 2);
  float*  part = (float*)take((size_t)BATCH * LN_PARTS * 2 * 4);
  float*  stats = (float*)take((size_t)BATCH * 2 * 4);
  half_t* ffn16 = q16;   // aliases q16..z16 (exactly 8192*2048 f16)

  // 1) convert inputs to f16
  launch_cvt(x,  x16,  TOT, stream);
  launch_cvt(Wq, wq16, DIM * DIM, stream);
  launch_cvt(Wk, wk16, DIM * DIM, stream);
  launch_cvt(Wv, wv16, DIM * DIM, stream);
  launch_cvt(Wo, wo16, DIM * DIM, stream);
  launch_cvt(W1, w1_16, DIM * DFF, stream);
  launch_cvt(W2, w2_16, DFF * DIM, stream);

  // 2) QKV projections (Q pre-scaled by 1/sqrt(DEPTH))
  launch_gemm(x16, wq16, bq, nullptr, nullptr, q16, M, DIM, DIM, 0.125f, 0, stream);
  launch_gemm(x16, wk16, bk, nullptr, nullptr, k16, M, DIM, DIM, 1.0f,   0, stream);
  launch_gemm(x16, wv16, bv, nullptr, nullptr, v16, M, DIM, DIM, 1.0f,   0, stream);

  // 3) flash attention -> z16 (already in (b, n, h*depth+d) layout)
  attn_kernel<<<dim3(BATCH * HEADS, SEQ_N / 128), 256, 0, stream>>>(q16, k16, v16, z16);

  // 4) output projection with fused residual: hbuf = x + (z @ Wo + bo)
  launch_gemm(z16, wo16, bo, x, hbuf, nullptr, M, DIM, DIM, 1.0f, 0, stream);

  // 5) LayerNorm 1 over (N, D) per batch -> out1 (f32) and out1_16
  ln_partial_kernel<<<dim3(LN_PARTS, BATCH), 256, 0, stream>>>(hbuf, part, PB);
  ln_finalize_kernel<<<BATCH, 32, 0, stream>>>(part, stats, 1.0f / (float)PB);
  ln_apply_kernel<<<TOT / 256, 256, 0, stream>>>(hbuf, g1, be1, stats, out1, out1_16, PB);

  // 6) FFN: ffn16 = relu(out1 @ W1 + b1); hbuf = out1 + relu(ffn16 @ W2 + b2)
  launch_gemm(out1_16, w1_16, b1, nullptr, nullptr, ffn16, M, DFF, DIM, 1.0f, 1, stream);
  launch_gemm(ffn16, w2_16, b2, out1, hbuf, nullptr, M, DIM, DFF, 1.0f, 1, stream);

  // 7) LayerNorm 2 -> d_out (f32)
  ln_partial_kernel<<<dim3(LN_PARTS, BATCH), 256, 0, stream>>>(hbuf, part, PB);
  ln_finalize_kernel<<<BATCH, 32, 0, stream>>>(part, stats, 1.0f / (float)PB);
  ln_apply_kernel<<<TOT / 256, 256, 0, stream>>>(hbuf, g2, be2, stats, (float*)d_out, nullptr, PB);
}
